// SetGNN_64914135712150
// MI455X (gfx1250) — matrix operand order repaired and checked by reading, not compiled, not run on hardware
//
#include <hip/hip_runtime.h>
#include <math.h>

typedef __attribute__((ext_vector_type(16))) _Float16 v16h;
typedef __attribute__((ext_vector_type(8)))  float    v8f;
typedef __attribute__((ext_vector_type(4)))  int      v4i;

#define N_NODES  100000
#define N_HEDGES 20000
#define N_INC    500000
#define HID      128
#define HEADS    4
#define NEG_SLOPE 0.2f
#define TM 128     // GEMM row tile
#define KP 132     // padded LDS K-stride (f16 elements)

// ---------------- small utility kernels ----------------
__global__ void k_fill(float* __restrict__ p, float v, int n) {
  int i = blockIdx.x * blockDim.x + threadIdx.x;
  if (i < n) p[i] = v;
}

__device__ inline float wave_sum(float v) {
#pragma unroll
  for (int o = 16; o > 0; o >>= 1) v += __shfl_xor(v, o, 32);
  return v;
}

__device__ inline void atomicMaxF(float* addr, float val) {
  unsigned* ua = (unsigned*)addr;
  unsigned old = __float_as_uint(*addr);
  while (__uint_as_float(old) < val) {
    unsigned assumed = old;
    old = atomicCAS(ua, assumed, __float_as_uint(val));
    if (old == assumed) break;
  }
}

// Convert W[128,N] (f32, row-major) -> Wt16[n*KP + k] (f16, transposed, zero-padded).
// Exactly the LDS image the GEMM wants, so workgroups can async-DMA it.
__global__ void k_wt16(const float* __restrict__ W, _Float16* __restrict__ out, int N) {
  int i = blockIdx.x * blockDim.x + threadIdx.x;
  if (i >= HID * KP) return;
  int n = i / KP, k = i % KP;
  float v = (k < HID && n < N) ? W[(size_t)k * N + n] : 0.f;
  out[i] = (_Float16)v;
}

// ---------------- WMMA GEMM: Y[M,N] = act(X[M,128] @ W[128,N] + b) ----------------
// NT = number of 16-wide N tiles (compile-time so accumulators stay in VGPRs).
// 256 threads = 8 waves; wave w computes rows [16w,16w+16) x all N columns.
// Weight tile arrives pre-transposed/padded in f16 and is staged into LDS with
// GLOBAL_LOAD_ASYNC_TO_LDS_B128 (ASYNCcnt) when the toolchain exposes it.
template <int NT>
__global__ __launch_bounds__(256)
void k_gemm128(const float* __restrict__ X, const _Float16* __restrict__ Wt16,
               const float* __restrict__ bias, float* __restrict__ Y,
               int M, int N, int relu)
{
  __shared__ _Float16 sX[TM * KP];     // X tile, row-major [r][k]
  __shared__ _Float16 sW[HID * KP];    // W transposed [n][k] (copied verbatim from Wt16)

  const int tid = threadIdx.x;
  const int rowbase = blockIdx.x * TM;

  __builtin_prefetch(X + (size_t)rowbase * HID, 0, 1);  // global_prefetch_b8

  // ---- async DMA of the weight image into LDS (16B per lane-op, ASYNCcnt) ----
#if __has_builtin(__builtin_amdgcn_global_load_async_to_lds_b128)
  {
    v4i* gsrc = (v4i*)Wt16;
    v4i* ldst = (v4i*)&sW[0];
    const int nchunk = (HID * KP * (int)sizeof(_Float16)) / 16;
    for (int i = tid; i < nchunk; i += 256) {
      __builtin_amdgcn_global_load_async_to_lds_b128(gsrc + i, ldst + i, 0, 0);
    }
  }
#else
  for (int i = tid; i < HID * KP; i += 256) sW[i] = Wt16[i];
#endif

  // ---- stage + convert the activation tile ----
  for (int idx = tid; idx < TM * HID; idx += 256) {
    int r = idx >> 7, k = idx & 127;
    int row = rowbase + r;
    sX[r * KP + k] = (row < M) ? (_Float16)X[(size_t)row * HID + k] : (_Float16)0.f;
  }

#if __has_builtin(__builtin_amdgcn_global_load_async_to_lds_b128)
#if __has_builtin(__builtin_amdgcn_s_wait_asynccnt)
  __builtin_amdgcn_s_wait_asynccnt(0);
#else
  asm volatile("s_wait_asynccnt 0x0" ::: "memory");
#endif
#endif
  __syncthreads();

  const int wave = tid >> 5, lane = tid & 31;
  const int half = lane >> 4, lm = lane & 15;
  const int m16 = wave * 16;

  v8f acc[NT] = {};

#pragma unroll
  for (int ks = 0; ks < 4; ++ks) {
    const int kb = ks * 32;
    // A fragment 16x32 f16 (ISA 7.12.2): lanes 0-15 hold K {0..7,16..23}, lanes 16-31 {8..15,24..31}
    v16h a;
    const _Float16* ax = &sX[(m16 + lm) * KP];
#pragma unroll
    for (int j = 0; j < 8; ++j) {
      int K0 = kb + 2 * j + 8 * half + ((j >= 4) ? 8 : 0);
      a[2 * j]     = ax[K0];
      a[2 * j + 1] = ax[K0 + 1];
    }
#pragma unroll
    for (int nt = 0; nt < NT; ++nt) {
      // B fragment 32x16: lane -> column N=lane%16, same K order as A: lanes 0-15 K {0..7,16..23}, lanes 16-31 {8..15,24..31}
      const _Float16* bx = &sW[(nt * 16 + lm) * KP + kb + 8 * half];
      v16h b;
#pragma unroll
      for (int j = 0; j < 8; ++j) { int o = 2 * j + ((j >= 4) ? 8 : 0); b[2 * j] = bx[o]; b[2 * j + 1] = bx[o + 1]; }
      acc[nt] = __builtin_amdgcn_wmma_f32_16x16x32_f16(
          false, a, false, b, (short)0, acc[nt], false, false);
    }
  }

  // C/D layout: VGPR r -> M = r + 8*(lane/16), N = lane%16
#pragma unroll
  for (int nt = 0; nt < NT; ++nt) {
    int col = nt * 16 + lm;
#pragma unroll
    for (int r = 0; r < 8; ++r) {
      int row = rowbase + m16 + r + 8 * half;
      if (row < M && col < N) {
        float v = acc[nt][r] + bias[col];
        if (relu) v = fmaxf(v, 0.f);
        Y[(size_t)row * N + col] = v;
      }
    }
  }
}

// ---------------- attention logits: alpha[s,h] = sum_c xK[s,h,c]*att_r[h,c] ----------------
__global__ void k_alpha(const float* __restrict__ xK, const float* __restrict__ attr,
                        float* __restrict__ alpha)
{
  int s = blockIdx.x;
  int t = threadIdx.x;  // 128 threads; wave h covers head h's 32 channels
  float v = xK[(size_t)s * HID + t] * attr[t];
  v = wave_sum(v);
  if ((t & 31) == 0) alpha[s * HEADS + (t >> 5)] = v;
}

// ---------------- segment softmax ----------------
__global__ void k_seg_max(const float* __restrict__ alpha, const int* __restrict__ src,
                          const int* __restrict__ dst, float* __restrict__ amax, int M4)
{
  int i = blockIdx.x * blockDim.x + threadIdx.x;
  if (i >= M4) return;
  int m = i >> 2, h = i & 3;
  float a = alpha[src[m] * HEADS + h];
  a = (a >= 0.f) ? a : NEG_SLOPE * a;
  atomicMaxF(&amax[dst[m] * HEADS + h], a);
}

__global__ void k_seg_exp(const float* __restrict__ alpha, const int* __restrict__ src,
                          const int* __restrict__ dst, const float* __restrict__ amax,
                          float* __restrict__ den, float* __restrict__ exw, int M4)
{
  int i = blockIdx.x * blockDim.x + threadIdx.x;
  if (i >= M4) return;
  int m = i >> 2, h = i & 3;
  float a = alpha[src[m] * HEADS + h];
  a = (a >= 0.f) ? a : NEG_SLOPE * a;
  int d = dst[m];
  float e = __expf(a - amax[d * HEADS + h]);
  exw[i] = e;
  atomicAdd(&den[d * HEADS + h], e);
}

// normalize incidence weights once: exw[m,h] /= max(den[dst,h], 1e-16)
__global__ void k_wnorm(float* __restrict__ exw, const int* __restrict__ dst,
                        const float* __restrict__ den, int M4)
{
  int i = blockIdx.x * blockDim.x + threadIdx.x;
  if (i >= M4) return;
  int m = i >> 2, h = i & 3;
  exw[i] = exw[i] / fmaxf(den[dst[m] * HEADS + h], 1e-16f);
}

// weighted scatter-add of value rows: float4 gathers, 4 atomics/thread (L2-resident)
__global__ void k_seg_scatter(const float* __restrict__ xV, const int* __restrict__ src,
                              const int* __restrict__ dst, const float* __restrict__ wgt,
                              float* __restrict__ tout, int M32)
{
  int i = blockIdx.x * blockDim.x + threadIdx.x;
  if (i >= M32) return;
  int m = i >> 5, q = i & 31;          // q indexes 4-float chunk within the 128-row
  int h = q >> 3;
  float w = wgt[m * HEADS + h];
  int s = src[m], d = dst[m];
  const float4 v = ((const float4*)(xV + (size_t)s * HID))[q];
  float* o = tout + (size_t)d * HID + q * 4;
  atomicAdd(o + 0, v.x * w);
  atomicAdd(o + 1, v.y * w);
  atomicAdd(o + 2, v.z * w);
  atomicAdd(o + 3, v.w * w);
}

// ---------------- fused residual + LayerNorm (wave per row) ----------------
// out = maybe_relu( LN( A + (B? maybe_relu(B):0) + (vec? vec:0) ) )
__global__ __launch_bounds__(256)
void k_ln_fused(const float* __restrict__ A, const float* __restrict__ B,
                const float* __restrict__ vec, const float* __restrict__ g,
                const float* __restrict__ bt, float* __restrict__ out,
                int M, int reluB, int reluOut)
{
  int wave = threadIdx.x >> 5, lane = threadIdx.x & 31;
  int row = blockIdx.x * 8 + wave;
  if (row >= M) return;
  float v[4];
#pragma unroll
  for (int i = 0; i < 4; ++i) {
    int e = lane + 32 * i;
    float x = A[(size_t)row * HID + e];
    if (B)  { float b = B[(size_t)row * HID + e]; x += reluB ? fmaxf(b, 0.f) : b; }
    if (vec) x += vec[e];
    v[i] = x;
  }
  float s = 0.f, q = 0.f;
#pragma unroll
  for (int i = 0; i < 4; ++i) { s += v[i]; q += v[i] * v[i]; }
  s = wave_sum(s); q = wave_sum(q);
  float mu  = s * (1.f / 128.f);
  float var = q * (1.f / 128.f) - mu * mu;
  float rstd = rsqrtf(var + 1e-5f);
#pragma unroll
  for (int i = 0; i < 4; ++i) {
    int e = lane + 32 * i;
    float y = (v[i] - mu) * rstd * g[e] + bt[e];
    if (reluOut) y = fmaxf(y, 0.f);
    out[(size_t)row * HID + e] = y;
  }
}

// ---------------- host-side orchestration ----------------
struct PmaP {
  const float *Kb, *Kw, *Vb, *Vw, *attr, *ln0b, *ln0g, *ln1b, *ln1g, *rb1, *rb2, *rw1, *rw2;
};

static PmaP getPma(void* const* d, int base) {
  PmaP p;
  p.Kb   = (const float*)d[base + 0];  p.Kw   = (const float*)d[base + 1];
  p.Vb   = (const float*)d[base + 2];  p.Vw   = (const float*)d[base + 3];
  p.attr = (const float*)d[base + 4];
  p.ln0b = (const float*)d[base + 5];  p.ln0g = (const float*)d[base + 6];
  p.ln1b = (const float*)d[base + 7];  p.ln1g = (const float*)d[base + 8];
  p.rb1  = (const float*)d[base + 9];  p.rb2  = (const float*)d[base + 10];
  p.rw1  = (const float*)d[base + 11]; p.rw2  = (const float*)d[base + 12];
  return p;
}

struct Ws {
  float *xK, *xV, *tout, *featA, *featB, *alpha, *amax, *den, *exw;
  _Float16* wt16;
};

static inline void cvt_w(const float* W, int N, const Ws& w, hipStream_t stream) {
  k_wt16<<<(HID * KP + 255) / 256, 256, 0, stream>>>(W, w.wt16, N);
}

static void run_pma(const PmaP& P, const float* xin, int S,
                    const int* src, const int* dst, int T,
                    const Ws& w, float* xout, hipStream_t stream)
{
  const int M4  = N_INC * HEADS;
  const int M32 = N_INC * 32;
  int gS = (S + TM - 1) / TM;
  int gT = (T + TM - 1) / TM;

  cvt_w(P.Kw, HID, w, stream);
  k_gemm128<8><<<gS, 256, 0, stream>>>(xin, w.wt16, P.Kb, w.xK, S, HID, 0);
  cvt_w(P.Vw, HID, w, stream);
  k_gemm128<8><<<gS, 256, 0, stream>>>(xin, w.wt16, P.Vb, w.xV, S, HID, 0);
  k_alpha<<<S, 128, 0, stream>>>(w.xK, P.attr, w.alpha);

  k_fill<<<(T * HEADS + 255) / 256, 256, 0, stream>>>(w.amax, -1e30f, T * HEADS);
  k_fill<<<(T * HEADS + 255) / 256, 256, 0, stream>>>(w.den, 0.f, T * HEADS);
  k_fill<<<(T * HID + 255) / 256, 256, 0, stream>>>(w.tout, 0.f, T * HID);

  k_seg_max<<<(M4 + 255) / 256, 256, 0, stream>>>(w.alpha, src, dst, w.amax, M4);
  k_seg_exp<<<(M4 + 255) / 256, 256, 0, stream>>>(w.alpha, src, dst, w.amax, w.den, w.exw, M4);
  k_wnorm<<<(M4 + 255) / 256, 256, 0, stream>>>(w.exw, dst, w.den, M4);
  k_seg_scatter<<<(M32 + 255) / 256, 256, 0, stream>>>(w.xV, src, dst, w.exw, w.tout, M32);

  // out = LN0(out + att_r)   (in-place)
  k_ln_fused<<<(T + 7) / 8, 256, 0, stream>>>(w.tout, nullptr, P.attr, P.ln0g, P.ln0b,
                                              w.tout, T, 0, 0);
  // rFF (reuse xK/xV as hidden/output)
  cvt_w(P.rw1, HID, w, stream);
  k_gemm128<8><<<gT, 256, 0, stream>>>(w.tout, w.wt16, P.rb1, w.xK, T, HID, 1);
  cvt_w(P.rw2, HID, w, stream);
  k_gemm128<8><<<gT, 256, 0, stream>>>(w.xK, w.wt16, P.rb2, w.xV, T, HID, 0);
  // x_out = relu(LN1(out + relu(rff)))
  k_ln_fused<<<(T + 7) / 8, 256, 0, stream>>>(w.tout, w.xV, nullptr, P.ln1g, P.ln1b,
                                              xout, T, 1, 1);
}

extern "C" void kernel_launch(void* const* d_in, const int* in_sizes, int n_in,
                              void* d_out, int out_size, void* d_ws, size_t ws_size,
                              hipStream_t stream) {
  (void)in_sizes; (void)n_in; (void)out_size; (void)ws_size;

  const float* x  = (const float*)d_in[0];
  const int*   ei = (const int*)d_in[1];      // [2, N_INC]
  const int* row0 = ei;                        // node ids
  const int* row1 = ei + N_INC;                // hyperedge ids
  // d_in[2] = norm (unused on the attention path)

  // params pytree flatten order (sorted dict keys): E2V[0], E2V[1], V2E[0], V2E[1], cls
  PmaP E2V0 = getPma(d_in, 3);
  PmaP E2V1 = getPma(d_in, 16);
  PmaP V2E0 = getPma(d_in, 29);
  PmaP V2E1 = getPma(d_in, 42);
  const float* cls_b1  = (const float*)d_in[55];
  const float* cls_b2  = (const float*)d_in[56];
  const float* cls_lnb = (const float*)d_in[57];
  const float* cls_lng = (const float*)d_in[58];
  const float* cls_w1  = (const float*)d_in[59];
  const float* cls_w2  = (const float*)d_in[60];

  // workspace layout (f32 units; wt16 region is 16B aligned)
  float* p = (float*)d_ws;
  Ws w;
  w.wt16  = (_Float16*)p; p += (HID * KP) / 2;          // 16896 f16 = 8448 f32
  w.xK    = p; p += (size_t)N_NODES * HID;
  w.xV    = p; p += (size_t)N_NODES * HID;
  w.tout  = p; p += (size_t)N_NODES * HID;
  w.featA = p; p += (size_t)N_NODES * HID;
  w.featB = p; p += (size_t)N_HEDGES * HID;
  w.alpha = p; p += (size_t)N_NODES * HEADS;
  w.amax  = p; p += (size_t)N_NODES * HEADS;
  w.den   = p; p += (size_t)N_NODES * HEADS;
  w.exw   = p; p += (size_t)N_INC * HEADS;

  // layer 0
  run_pma(V2E0, x,        N_NODES,  row0, row1, N_HEDGES, w, w.featB, stream);
  run_pma(E2V0, w.featB,  N_HEDGES, row1, row0, N_NODES,  w, w.featA, stream);
  // layer 1
  run_pma(V2E1, w.featA,  N_NODES,  row0, row1, N_HEDGES, w, w.featB, stream);
  run_pma(E2V1, w.featB,  N_HEDGES, row1, row0, N_NODES,  w, w.featA, stream);

  // classifier: h = LN(relu(x@w1+b1)); out = h@w2+b2
  int gV = (N_NODES + TM - 1) / TM;
  cvt_w(cls_w1, HID, w, stream);
  k_gemm128<8><<<gV, 256, 0, stream>>>(w.featA, w.wt16, cls_b1, w.xK, N_NODES, HID, 1);
  k_ln_fused<<<(N_NODES + 7) / 8, 256, 0, stream>>>(w.xK, nullptr, nullptr, cls_lng, cls_lnb,
                                                    w.xK, N_NODES, 0, 0);
  cvt_w(cls_w2, 40, w, stream);
  k_gemm128<3><<<gV, 256, 0, stream>>>(w.xK, w.wt16, cls_b2, (float*)d_out, N_NODES, 40, 0);
}
